// Qwen3NextGatedDeltaNet_76089640616332
// MI455X (gfx1250) — matrix operand order repaired and checked by reading, not compiled, not run on hardware
//
#include <hip/hip_runtime.h>
#include <hip/hip_bf16.h>
#include <stdint.h>

#define L_SEQ   4096
#define HID     2048
#define HKH     16
#define HVH     32
#define DKD     128
#define DVD     128
#define QKVZ_N  12288   // 2*2048 + 2*4096
#define CONV_D  8192    // 2*2048 + 4096
#define VALD    4096

typedef __bf16 bf16_t;
typedef __attribute__((ext_vector_type(16))) __bf16 v16bf;
typedef __attribute__((ext_vector_type(8)))  float  v8f;

union FragBF { v16bf v; unsigned u[8]; };

// exact parameter types of __builtin_amdgcn_global_load_async_to_lds_b128
// (from hipcc diagnostic: "__vector_size__(4 * sizeof(int)) int __device__ *")
typedef int v4i_t __attribute__((__vector_size__(16)));
typedef __attribute__((address_space(1))) v4i_t as1_v4i;
typedef __attribute__((address_space(3))) v4i_t as3_v4i;

// ------- CDNA5 async global->LDS copy (guarded; fallback = reg prefetch) ----
#if defined(__has_builtin)
# if __has_builtin(__builtin_amdgcn_global_load_async_to_lds_b128)
#  define HAVE_ASYNC_LDS 1
# endif
# if __has_builtin(__builtin_amdgcn_s_wait_asynccnt)
#  define HAVE_WAIT_ASYNC 1
# endif
#endif
#ifndef HAVE_ASYNC_LDS
# define HAVE_ASYNC_LDS 0
#endif
#ifndef HAVE_WAIT_ASYNC
# define HAVE_WAIT_ASYNC 0
#endif

__device__ __forceinline__ void wait_asynccnt0() {
#if HAVE_WAIT_ASYNC
  __builtin_amdgcn_s_wait_asynccnt(0);
#else
  asm volatile("s_wait_asynccnt 0x0" ::: "memory");
#endif
}

__device__ __forceinline__ float siluf(float x) { return x / (1.f + __expf(-x)); }

// ---------------------------------------------------------------- f32 -> bf16
__global__ void cvt_f32_bf16(const float* __restrict__ in, bf16_t* __restrict__ out, int n4) {
  int i = blockIdx.x * blockDim.x + threadIdx.x;
  if (i < n4) {
    float4 f = ((const float4*)in)[i];
    bf16_t o[4] = {(bf16_t)f.x, (bf16_t)f.y, (bf16_t)f.z, (bf16_t)f.w};
    *(uint2*)(out + 4 * i) = *(uint2*)o;
  }
}

// ------------------------------------------------- bf16 WMMA GEMM, C = A @ B
// A[M,K] rm bf16, B[K,N] rm bf16, C[M,N] f32.  M,N % 128 == 0, K % 32 == 0.
// Block = 256 thr (8 waves); BM=BN=128, BK=32; wave -> 64x32 (8 accumulators).
// Double-buffered LDS; A tile via async global->LDS copy when available.
#define BM 128
#define BN 128
#define BK 32
#define LDSK 40   // padded LDS row stride (bf16 elems); 80 B, 16B-aligned rows

#if HAVE_ASYNC_LDS
#define ISSUE_A(Ab, AsBuf) do { _Pragma("unroll")                               \
  for (int j_ = 0; j_ < 2; ++j_) { int m_ = a_m + j_ * 64;                      \
    __builtin_amdgcn_global_load_async_to_lds_b128(                             \
        (as1_v4i*)(void*)((Ab) + (int64_t)m_ * K + a_kc),                       \
        (as3_v4i*)(void*)((AsBuf) + m_ * LDSK + a_kc), 0, 0); } } while (0)
#define STORE_A(AsBuf) ((void)0)
#define WAIT_A() wait_asynccnt0()
#else
#define ISSUE_A(Ab, AsBuf) do { _Pragma("unroll")                               \
  for (int j_ = 0; j_ < 2; ++j_) { int m_ = a_m + j_ * 64;                      \
    aregs[j_] = *(const uint4*)((Ab) + (int64_t)m_ * K + a_kc); } } while (0)
#define STORE_A(AsBuf) do { _Pragma("unroll")                                   \
  for (int j_ = 0; j_ < 2; ++j_) { int m_ = a_m + j_ * 64;                      \
    *(uint4*)((AsBuf) + m_ * LDSK + a_kc) = aregs[j_]; } } while (0)
#define WAIT_A() ((void)0)
#endif

#define LOAD_B(Bb) do { _Pragma("unroll")                                       \
  for (int j_ = 0; j_ < 2; ++j_) { int kk_ = b_kk + j_ * 16;                    \
    bregs[j_] = *(const uint4*)((Bb) + (int64_t)kk_ * N + b_nn); } } while (0)
#define STORE_B(BsBuf) do { _Pragma("unroll")                                   \
  for (int j_ = 0; j_ < 2; ++j_) { int kk_ = b_kk + j_ * 16;                    \
    const bf16_t* e_ = (const bf16_t*)&bregs[j_]; _Pragma("unroll")             \
    for (int t_ = 0; t_ < 8; ++t_)                                              \
      (BsBuf)[(b_nn + t_) * LDSK + kk_] = e_[t_]; } } while (0)

__launch_bounds__(256, 2)
__global__ void gemm_bf16_wmma(const bf16_t* __restrict__ A, const bf16_t* __restrict__ B,
                               float* __restrict__ C, int M, int N, int K) {
  __shared__ __align__(16) bf16_t As[2][BM * LDSK];
  __shared__ __align__(16) bf16_t Bs[2][BN * LDSK];   // transposed: Bs[n][k]
  const int tid   = threadIdx.x;
  const int lane  = tid & 31;
  const int wave  = tid >> 5;
  const int wr    = wave >> 2;        // 0..1
  const int wc    = wave & 3;         // 0..3
  const int m0    = blockIdx.y * BM;
  const int n0    = blockIdx.x * BN;
  const int laneM = lane & 15;
  const int grp   = lane >> 4;
  // tile-copy coordinates (8 bf16 = 16 B per chunk, 2 chunks/thread/tile)
  const int a_m  = tid >> 2;          // + j*64
  const int a_kc = (tid & 3) * 8;
  const int b_kk = tid >> 4;          // + j*16
  const int b_nn = (tid & 15) * 8;

  const v8f vzero = {0.f, 0.f, 0.f, 0.f, 0.f, 0.f, 0.f, 0.f};
  v8f acc[4][2];
#pragma unroll
  for (int i = 0; i < 4; ++i) { acc[i][0] = vzero; acc[i][1] = vzero; }

  uint4 bregs[2];
#if !HAVE_ASYNC_LDS
  uint4 aregs[2];
#endif

  // prologue: stage slab 0 into buffer 0
  {
    const bf16_t* Ab = A + (int64_t)m0 * K;
    const bf16_t* Bb = B + n0;
    ISSUE_A(Ab, As[0]);
    LOAD_B(Bb);
    STORE_A(As[0]);
    STORE_B(Bs[0]);
    WAIT_A();
  }
  __syncthreads();

  const int NK = K / BK;
  for (int i = 0; i < NK; ++i) {
    const int cur = i & 1;
    const int nxt = cur ^ 1;
    const bool more = (i + 1 < NK);
    if (more) {                       // start fetching slab i+1
      const bf16_t* Ab = A + (int64_t)m0 * K + (i + 1) * BK;
      const bf16_t* Bb = B + (int64_t)(i + 1) * BK * N + n0;
      ISSUE_A(Ab, As[nxt]);
      LOAD_B(Bb);
    }
    // ---- compute on slab i ----
    const bf16_t* Asc = As[cur];
    const bf16_t* Bsc = Bs[cur];
    FragBF bfr[2];
#pragma unroll
    for (int nt = 0; nt < 2; ++nt) {  // B frag: k = grp*16 + v*2
      const bf16_t* brow = &Bsc[(wc * 32 + nt * 16 + laneM) * LDSK];
#pragma unroll
      for (int v = 0; v < 8; ++v)
        bfr[nt].u[v] = *(const unsigned*)(brow + grp * 16 + v * 2);
    }
#pragma unroll
    for (int mt = 0; mt < 4; ++mt) {  // A frag: k = grp*16 + v*2
      FragBF afr;
      const bf16_t* arow = &Asc[(wr * 64 + mt * 16 + laneM) * LDSK];
#pragma unroll
      for (int v = 0; v < 8; ++v)
        afr.u[v] = *(const unsigned*)(arow + grp * 16 + v * 2);
      acc[mt][0] = __builtin_amdgcn_wmma_f32_16x16x32_bf16(false, afr.v, false, bfr[0].v,
                                                           (short)0, acc[mt][0], false, false);
      acc[mt][1] = __builtin_amdgcn_wmma_f32_16x16x32_bf16(false, afr.v, false, bfr[1].v,
                                                           (short)0, acc[mt][1], false, false);
    }
    if (more) {
      STORE_A(As[nxt]);
      STORE_B(Bs[nxt]);
      WAIT_A();
    }
    __syncthreads();
  }
  // C/D layout: lanes0-15 vgpr r -> M=r, lanes16-31 -> M=8+r; N = lane&15
#pragma unroll
  for (int mt = 0; mt < 4; ++mt)
#pragma unroll
    for (int nt = 0; nt < 2; ++nt) {
      float* cp = C + (int64_t)(m0 + wr * 64 + mt * 16 + grp * 8) * N
                    + (n0 + wc * 32 + nt * 16 + laneM);
#pragma unroll
      for (int r = 0; r < 8; ++r) { *cp = acc[mt][nt][r]; cp += N; }
    }
}
#undef ISSUE_A
#undef STORE_A
#undef WAIT_A
#undef LOAD_B
#undef STORE_B

// ----------------------------------------------------- skinny GEMM  ba = H @ W_ba
__global__ void gemm_ba(const float* __restrict__ H, const float* __restrict__ Wba,
                        float* __restrict__ out) {
  int t = blockIdx.x, j = threadIdx.x;      // block = 64
  float s = 0.f;
  const float* hrow = H + (int64_t)t * HID;
  for (int k = 0; k < HID; ++k) s += hrow[k] * Wba[k * 64 + j];
  out[t * 64 + j] = s;
}

// -------------------------------------------- causal conv (K=4) + SiLU, gather
__global__ void conv_silu(const float* __restrict__ qkvz, const float* __restrict__ convw,
                          float* __restrict__ out) {
  int c = blockIdx.x * blockDim.x + threadIdx.x;   // 0..8191
  int t = blockIdx.y;
  int col;
  if (c < 2048)      { int hh = c >> 7;  col = hh * 768 + (c & 127); }          // q
  else if (c < 4096) { int c2 = c - 2048; int hh = c2 >> 7; col = hh * 768 + 128 + (c2 & 127); } // k
  else { int c3 = c - 4096; int vh = c3 >> 7; col = (vh >> 1) * 768 + 256 + (vh & 1) * 128 + (c3 & 127); } // v
  float acc = 0.f;
#pragma unroll
  for (int j = 0; j < 4; ++j) {
    int tt = t + j - 3;
    float x = (tt >= 0) ? qkvz[(int64_t)tt * QKVZ_N + col] : 0.f;
    acc += x * convw[c * 4 + j];
  }
  out[(int64_t)t * CONV_D + c] = siluf(acc);
}

// ------------------------------- L2-normalize q (with 1/sqrt(DK)) and k in place
__global__ void l2norm_qk(float* __restrict__ conv) {
  int t = blockIdx.y, b = blockIdx.x;       // b: 0..15 q heads, 16..31 k heads
  int isK = b >> 4, hh = b & 15, d = threadIdx.x;  // 128 threads
  float* ptr = conv + (int64_t)t * CONV_D + isK * 2048 + hh * 128;
  float x = ptr[d];
  __shared__ float red[128];
  red[d] = x * x; __syncthreads();
  for (int s = 64; s > 0; s >>= 1) { if (d < s) red[d] += red[d + s]; __syncthreads(); }
  float inv = rsqrtf(red[0] + 1e-6f);
  if (!isK) inv *= 0.08838834764831845f;    // DK^-0.5
  ptr[d] = x * inv;
}

// ---------------------------------------------------------------- g and beta
__global__ void gbeta_kernel(const float* __restrict__ ba, const float* __restrict__ A_log,
                             const float* __restrict__ dt_bias, float* __restrict__ g,
                             float* __restrict__ beta) {
  int i = blockIdx.x * blockDim.x + threadIdx.x;
  if (i >= L_SEQ * HVH) return;
  int t = i >> 5, vh = i & 31, hh = vh >> 1, r = vh & 1;
  float b = ba[t * 64 + hh * 4 + r];
  float a = ba[t * 64 + hh * 4 + 2 + r];
  float x = a + dt_bias[vh];
  float sp = (x > 20.f) ? x : log1pf(__expf(x));
  g[i]    = -__expf(A_log[vh]) * sp;
  beta[i] = 1.f / (1.f + __expf(-b));
}

// --------------------------------------- sequential delta-rule scan, 1 blk/head
__launch_bounds__(256, 1)
__global__ void scan_kernel(const float* __restrict__ conv, const float* __restrict__ g,
                            const float* __restrict__ beta, float* __restrict__ core) {
  const int vh = blockIdx.x;                 // 0..31
  const int hh = vh >> 1;
  const int tid = threadIdx.x;
  const int v = tid & 127;                   // state column (DV)
  const int h = tid >> 7;                    // K half: 0 -> k 0..63, 1 -> k 64..127

  float S[64];
#pragma unroll
  for (int j = 0; j < 64; ++j) S[j] = 0.f;

  __shared__ float kbuf[128], qbuf[128], dbuf[128], partial[256];
  const float* qbase = conv + hh * 128;
  const float* kbase = conv + 2048 + hh * 128;
  const float* vbase = conv + 4096 + vh * 128;

  for (int t = 0; t < L_SEQ; ++t) {
    int64_t roff = (int64_t)t * CONV_D;
    if (h == 0) kbuf[v] = kbase[roff + v];
    else        qbuf[v] = qbase[roff + v];
    __syncthreads();

    float dec = __expf(g[t * 32 + vh]);
    float p = 0.f;
#pragma unroll
    for (int j = 0; j < 64; ++j) {
      float kk = kbuf[h * 64 + j];
      S[j] *= dec;
      p += kk * S[j];
    }
    partial[tid] = p;
    __syncthreads();
    if (h == 0) {
      float pred = partial[v] + partial[v + 128];
      float vt   = vbase[roff + v];
      dbuf[v] = (vt - pred) * beta[t * 32 + vh];
    }
    __syncthreads();

    float d = dbuf[v];
    float o = 0.f;
#pragma unroll
    for (int j = 0; j < 64; ++j) {
      float kk = kbuf[h * 64 + j];
      S[j] += kk * d;
      o += qbuf[h * 64 + j] * S[j];
    }
    partial[tid] = o;
    __syncthreads();
    if (h == 0)
      core[(int64_t)t * VALD + vh * 128 + v] = partial[v] + partial[v + 128];
  }
}

// --------------------------------------- RMS-norm * weight * SiLU(z) -> bf16
__global__ void norm_gate(const float* __restrict__ core, const float* __restrict__ qkvz,
                          const float* __restrict__ nw, bf16_t* __restrict__ ybf) {
  int t = blockIdx.y, vh = blockIdx.x, d = threadIdx.x;   // 128 threads
  int hh = vh >> 1, r = vh & 1;
  float x = core[(int64_t)t * VALD + vh * 128 + d];
  __shared__ float red[128];
  red[d] = x * x; __syncthreads();
  for (int s = 64; s > 0; s >>= 1) { if (d < s) red[d] += red[d + s]; __syncthreads(); }
  float inv = rsqrtf(red[0] * (1.f / 128.f) + 1e-6f);
  float z = qkvz[(int64_t)t * QKVZ_N + hh * 768 + 512 + r * 128 + d];
  float y = x * inv * nw[d] * siluf(z);
  ybf[(int64_t)t * VALD + vh * 128 + d] = (bf16_t)y;
}

// ============================================================================
extern "C" void kernel_launch(void* const* d_in, const int* in_sizes, int n_in,
                              void* d_out, int out_size, void* d_ws, size_t ws_size,
                              hipStream_t stream) {
  const float* hidden  = (const float*)d_in[0];
  const float* Wqkvz   = (const float*)d_in[1];
  const float* Wba     = (const float*)d_in[2];
  const float* convw   = (const float*)d_in[3];
  const float* A_log   = (const float*)d_in[4];
  const float* dt_bias = (const float*)d_in[5];
  const float* norm_w  = (const float*)d_in[6];
  const float* Wout    = (const float*)d_in[7];
  float* out = (float*)d_out;

  char* ws = (char*)d_ws;
  size_t off = 0;
  auto alloc = [&](size_t bytes) -> void* {
    void* p = ws + off;
    off += (bytes + 255) & ~(size_t)255;
    return p;
  };
  bf16_t* hid_b   = (bf16_t*)alloc((size_t)L_SEQ * HID * 2);
  bf16_t* wqkvz_b = (bf16_t*)alloc((size_t)HID * QKVZ_N * 2);
  bf16_t* wout_b  = (bf16_t*)alloc((size_t)VALD * HID * 2);
  float*  qkvz    = (float*) alloc((size_t)L_SEQ * QKVZ_N * 4);
  float*  ba      = (float*) alloc((size_t)L_SEQ * 64 * 4);
  float*  conv    = (float*) alloc((size_t)L_SEQ * CONV_D * 4);
  float*  g       = (float*) alloc((size_t)L_SEQ * HVH * 4);
  float*  beta    = (float*) alloc((size_t)L_SEQ * HVH * 4);
  float*  core    = (float*) alloc((size_t)L_SEQ * VALD * 4);
  bf16_t* y_b     = (bf16_t*)alloc((size_t)L_SEQ * VALD * 2);

  // 1) bf16 conversions
  {
    int n4 = L_SEQ * HID / 4;
    cvt_f32_bf16<<<(n4 + 255) / 256, 256, 0, stream>>>(hidden, hid_b, n4);
    n4 = HID * QKVZ_N / 4;
    cvt_f32_bf16<<<(n4 + 255) / 256, 256, 0, stream>>>(Wqkvz, wqkvz_b, n4);
    n4 = VALD * HID / 4;
    cvt_f32_bf16<<<(n4 + 255) / 256, 256, 0, stream>>>(Wout, wout_b, n4);
  }
  // 2) qkvz = H @ W_qkvz   (4096 x 2048 x 12288, WMMA)
  gemm_bf16_wmma<<<dim3(QKVZ_N / BN, L_SEQ / BM), 256, 0, stream>>>(
      hid_b, wqkvz_b, qkvz, L_SEQ, QKVZ_N, HID);
  // 3) ba = H @ W_ba
  gemm_ba<<<L_SEQ, 64, 0, stream>>>(hidden, Wba, ba);
  // 4) causal conv + SiLU
  conv_silu<<<dim3(CONV_D / 256, L_SEQ), 256, 0, stream>>>(qkvz, convw, conv);
  // 5) normalize q (with DK^-0.5) and k in place
  l2norm_qk<<<dim3(32, L_SEQ), 128, 0, stream>>>(conv);
  // 6) g, beta
  gbeta_kernel<<<(L_SEQ * HVH) / 256, 256, 0, stream>>>(ba, A_log, dt_bias, g, beta);
  // 7) sequential delta-rule scan (32 heads in parallel)
  scan_kernel<<<HVH, 256, 0, stream>>>(conv, g, beta, core);
  // 8) RMS norm * gate -> bf16
  norm_gate<<<dim3(HVH, L_SEQ), 128, 0, stream>>>(core, qkvz, norm_w, y_b);
  // 9) out = y @ W_out   (4096 x 4096 x 2048, WMMA)
  gemm_bf16_wmma<<<dim3(HID / BN, L_SEQ / BM), 256, 0, stream>>>(
      y_b, wout_b, out, L_SEQ, HID, VALD);
}
